// EncoderUnit_27066883900004
// MI455X (gfx1250) — compile-verified, matrix operand order checked by reading, not run on hardware
//
#include <hip/hip_runtime.h>
#include <hip/hip_bf16.h>

// ---------------------------------------------------------------------------
// Types for CDNA5 WMMA (wave32, 16x16x32 bf16 -> f32)
// ---------------------------------------------------------------------------
typedef __bf16 bf16_t;
typedef __attribute__((ext_vector_type(16))) __bf16 v16bf;
typedef __attribute__((ext_vector_type(8)))  __bf16 v8bf;
typedef __attribute__((ext_vector_type(8)))  float  v8f;
typedef int v4i __attribute__((vector_size(16)));

union Frag16 { v16bf v; v8bf h[2]; };

static constexpr int   Bv = 4, Sv = 2048, Ev = 1024, Hv = 16, HDv = 64, HIDv = 4096;
static constexpr float EPSv = 1e-6f;

// ---------------------------------------------------------------------------
// Async global->LDS copy path (CDNA5). Falls back to direct global fragment
// loads if the builtin is unavailable on this toolchain.
// ---------------------------------------------------------------------------
#if defined(__gfx1250__) && __has_builtin(__builtin_amdgcn_global_load_async_to_lds_b128)
#define USE_ASYNC_LDS 1
#pragma message("ASYNC_LDS: using __builtin_amdgcn_global_load_async_to_lds_b128")
#else
#define USE_ASYNC_LDS 0
#pragma message("ASYNC_LDS: builtin not available, using direct-global fragment path")
#endif

#if USE_ASYNC_LDS
__device__ __forceinline__ void async_copy_b128(const void* gsrc, void* ldst) {
  __builtin_amdgcn_global_load_async_to_lds_b128(
      (__attribute__((address_space(1))) v4i*)gsrc,
      (__attribute__((address_space(3))) v4i*)ldst,
      0, 0);
}
#endif

// ---------------------------------------------------------------------------
// fp32 -> bf16 conversion (grid-stride)
// ---------------------------------------------------------------------------
__global__ void cvt_f32_to_bf16(const float* __restrict__ in, bf16_t* __restrict__ out, long n) {
  long i = (long)blockIdx.x * blockDim.x + threadIdx.x;
  const long stride = (long)gridDim.x * blockDim.x;
  for (; i < n; i += stride) out[i] = (bf16_t)in[i];
}

// ---------------------------------------------------------------------------
// Generic GEMM: C[M,N] = act(A[M,K] * W[N,K]^T + bias[N])
// A, W bf16 row-major. Block = 256 thr = 8 waves; block tile 128x128;
// wave tile 64(M) x 32(N) = 4x2 WMMA accumulators; k-step 32.
//
// Async path: A/W k-slices staged to LDS with global_load_async_to_lds_b128,
// double buffered; per-wave s_wait_asynccnt + workgroup barrier, then WMMA
// fragments are read from LDS (padded stride 40 elems = 80B, 16B chunks).
//
// 16-bit A-fragment layout (ISA 7.12.2): lane l holds row M=l&15;
// bf16 elems i=0..7 -> K = ((l>>4)*8) + i, i=8..15 -> K = 16 + ((l>>4)*8) + (i-8).
// W[N,K] rows index K the same way, so B fragments use the identical pattern.
// ---------------------------------------------------------------------------
template <bool RELU, bool OUT_BF16>
__global__ __launch_bounds__(256) void gemm_bf16_wmma(
    const bf16_t* __restrict__ A, const bf16_t* __restrict__ W,
    const float* __restrict__ bias, void* __restrict__ Cv,
    int M, int N, int K) {
  const int tid  = threadIdx.x;
  const int lane = tid & 31;
  const int wave = tid >> 5;
  const int g    = lane >> 4;   // lane half
  const int r    = lane & 15;
  const int m0b  = blockIdx.y * 128;
  const int n0b  = blockIdx.x * 128;
  const int wm   = (wave >> 2) * 64;   // wave M offset within block tile
  const int wn   = (wave & 3) * 32;    // wave N offset within block tile
  const int koff = g * 8;

  const v8f vzero = {};
  v8f acc[4][2];
#pragma unroll
  for (int i = 0; i < 4; ++i)
#pragma unroll
    for (int j = 0; j < 2; ++j) acc[i][j] = vzero;

#if USE_ASYNC_LDS
  constexpr int LDK = 40;  // padded LDS row stride in elems (80B, multiple of 16B)
  __shared__ alignas(16) bf16_t lA[2][128 * LDK];
  __shared__ alignas(16) bf16_t lW[2][128 * LDK];

  // staging: 128 rows x 32 cols bf16 = 8KB per matrix = 512 x 16B chunks;
  // 256 threads x 2 chunks each. chunk -> row = srow + i*64, col = scol.
  const int srow = tid >> 2;         // 0..63
  const int scol = (tid & 3) * 8;    // 0,8,16,24 (elements)

  auto stage = [&](int buf, int k0) {
#pragma unroll
    for (int i = 0; i < 2; ++i) {
      const int row = srow + i * 64;
      async_copy_b128(A + (size_t)(m0b + row) * K + k0 + scol,
                      &lA[buf][row * LDK + scol]);
      async_copy_b128(W + (size_t)(n0b + row) * K + k0 + scol,
                      &lW[buf][row * LDK + scol]);
    }
  };

  stage(0, 0);
  int buf = 0;
  for (int k0 = 0; k0 < K; k0 += 32) {
    const bool more = (k0 + 32) < K;
    if (more) stage(buf ^ 1, k0 + 32);  // prefetch next slice (4 more async ops)
    // wait for the 4 async ops that filled the current buffer (in-order completion)
    if (more) asm volatile("s_wait_asynccnt 0x4" ::: "memory");
    else      asm volatile("s_wait_asynccnt 0x0" ::: "memory");
    __syncthreads();

    v16bf af[4], bfr[2];
#pragma unroll
    for (int mi = 0; mi < 4; ++mi) {
      const bf16_t* p = &lA[buf][(wm + mi * 16 + r) * LDK];
      Frag16 f;
      f.h[0] = *(const v8bf*)(p + koff);
      f.h[1] = *(const v8bf*)(p + 16 + koff);
      af[mi] = f.v;
    }
#pragma unroll
    for (int ni = 0; ni < 2; ++ni) {
      const bf16_t* p = &lW[buf][(wn + ni * 16 + r) * LDK];
      Frag16 f;
      f.h[0] = *(const v8bf*)(p + koff);
      f.h[1] = *(const v8bf*)(p + 16 + koff);
      bfr[ni] = f.v;
    }
#pragma unroll
    for (int mi = 0; mi < 4; ++mi)
#pragma unroll
      for (int ni = 0; ni < 2; ++ni)
        acc[mi][ni] = __builtin_amdgcn_wmma_f32_16x16x32_bf16(
            false, af[mi], false, bfr[ni], (short)0, acc[mi][ni], false, false);

    __syncthreads();  // all waves done reading buf before it is refilled
    buf ^= 1;
  }
#else
  // Fallback: fragments straight from global memory.
  const bf16_t* arow[4];
  const bf16_t* wrow[2];
#pragma unroll
  for (int mi = 0; mi < 4; ++mi) arow[mi] = A + (size_t)(m0b + wm + mi * 16 + r) * K;
#pragma unroll
  for (int ni = 0; ni < 2; ++ni) wrow[ni] = W + (size_t)(n0b + wn + ni * 16 + r) * K;

  for (int k = 0; k < K; k += 32) {
    v16bf af[4], bfr[2];
#pragma unroll
    for (int mi = 0; mi < 4; ++mi) {
      Frag16 f;
      f.h[0] = *(const v8bf*)(arow[mi] + k + koff);
      f.h[1] = *(const v8bf*)(arow[mi] + k + 16 + koff);
      af[mi] = f.v;
    }
#pragma unroll
    for (int ni = 0; ni < 2; ++ni) {
      Frag16 f;
      f.h[0] = *(const v8bf*)(wrow[ni] + k + koff);
      f.h[1] = *(const v8bf*)(wrow[ni] + k + 16 + koff);
      bfr[ni] = f.v;
    }
#pragma unroll
    for (int mi = 0; mi < 4; ++mi)
#pragma unroll
      for (int ni = 0; ni < 2; ++ni)
        acc[mi][ni] = __builtin_amdgcn_wmma_f32_16x16x32_bf16(
            false, af[mi], false, bfr[ni], (short)0, acc[mi][ni], false, false);
  }
#endif

  // Epilogue. D fragment: lane holds column n = base + (lane&15),
  // VGPR rr holds row m = base + rr + 8*(lane>>4).
#pragma unroll
  for (int mi = 0; mi < 4; ++mi)
#pragma unroll
    for (int ni = 0; ni < 2; ++ni) {
      const int n   = n0b + wn + ni * 16 + r;
      const float b = bias[n];
#pragma unroll
      for (int rr = 0; rr < 8; ++rr) {
        const int m = m0b + wm + mi * 16 + rr + 8 * g;
        float v = acc[mi][ni][rr] + b;
        if (RELU) v = fmaxf(v, 0.0f);
        if (OUT_BF16) ((bf16_t*)Cv)[(size_t)m * N + n] = (bf16_t)v;
        else          ((float*)Cv)[(size_t)m * N + n] = v;
      }
    }
}

// ---------------------------------------------------------------------------
// Split fused QKV [B,S,3E] fp32 into bf16 Q [BH,S,HD], K [BH,S,HD],
// and V transposed [BH,HD,S] (so P·V B-fragments read contiguously).
// ---------------------------------------------------------------------------
__global__ void split_qkv_kernel(const float* __restrict__ qkv, bf16_t* __restrict__ qb,
                                 bf16_t* __restrict__ kb, bf16_t* __restrict__ vtb) {
  const long n = (long)Bv * Sv * 3 * Ev;
  long i = (long)blockIdx.x * blockDim.x + threadIdx.x;
  const long stride = (long)gridDim.x * blockDim.x;
  for (; i < n; i += stride) {
    const long rowc = i / (3 * Ev);
    const int  c    = (int)(i - rowc * (3 * Ev));
    const int  b    = (int)(rowc / Sv);
    const int  s    = (int)(rowc - (long)b * Sv);
    const int  h    = c / (3 * HDv);
    const int  rem  = c - h * (3 * HDv);
    const int  wch  = rem / HDv;
    const int  d    = rem - wch * HDv;
    const long bh   = (long)b * Hv + h;
    const bf16_t val = (bf16_t)qkv[i];
    if (wch == 0)      qb [(bh * Sv + s) * HDv + d] = val;
    else if (wch == 1) kb [(bh * Sv + s) * HDv + d] = val;
    else               vtb[(bh * HDv + d) * Sv + s] = val;
  }
}

// ---------------------------------------------------------------------------
// Flash-style attention. grid = (S/64, B*H), block = 128 (4 waves).
// Each wave handles one 16-query tile: online softmax over 64 key-steps of 32.
// Per step: 4 WMMAs for scores (two 16x16 n-tiles, k=64 in 2 chunks) and
// 4 WMMAs for P·V (ctx 16x64). P is reshaped D-frag -> A-frag through LDS.
// ---------------------------------------------------------------------------
__global__ __launch_bounds__(128) void attention_kernel(
    const bf16_t* __restrict__ qb, const bf16_t* __restrict__ kb,
    const bf16_t* __restrict__ vtb, const float* __restrict__ mask,
    bf16_t* __restrict__ ctxb) {
  __shared__ alignas(16) bf16_t P[4][16][32];  // per-wave 16x32 probability tile

  const int lane = threadIdx.x & 31;
  const int wave = threadIdx.x >> 5;
  const int g    = lane >> 4;
  const int r    = lane & 15;
  const int bh   = blockIdx.y;
  const int b    = bh / Hv;
  const int h    = bh - b * Hv;
  const int q0   = (blockIdx.x * 4 + wave) * 16;
  const int koff = g * 8;

  // Q fragments (row m = q0 + r, K = head dim, two 32-wide chunks)
  const bf16_t* qrow = qb + ((size_t)bh * Sv + (q0 + r)) * HDv;
  v16bf aq[2];
#pragma unroll
  for (int kk = 0; kk < 2; ++kk) {
    Frag16 f;
    f.h[0] = *(const v8bf*)(qrow + kk * 32 + koff);
    f.h[1] = *(const v8bf*)(qrow + kk * 32 + 16 + koff);
    aq[kk] = f.v;
  }

  const v8f vzero = {};
  v8f ctx[4];
#pragma unroll
  for (int ni = 0; ni < 4; ++ni) ctx[ni] = vzero;
  float mrow[8], lrow[8];
#pragma unroll
  for (int rr = 0; rr < 8; ++rr) { mrow[rr] = -1e30f; lrow[rr] = 0.0f; }

  for (int t0 = 0; t0 < Sv; t0 += 32) {
    // ---- scores = Q · K^T (K stored [t, d] row-major) ----
    v8f s0 = vzero, s1 = vzero;
    const bf16_t* k0row = kb + ((size_t)bh * Sv + (t0 + r)) * HDv;
    const bf16_t* k1row = kb + ((size_t)bh * Sv + (t0 + 16 + r)) * HDv;
#pragma unroll
    for (int kk = 0; kk < 2; ++kk) {
      Frag16 f0, f1;
      f0.h[0] = *(const v8bf*)(k0row + kk * 32 + koff);
      f0.h[1] = *(const v8bf*)(k0row + kk * 32 + 16 + koff);
      f1.h[0] = *(const v8bf*)(k1row + kk * 32 + koff);
      f1.h[1] = *(const v8bf*)(k1row + kk * 32 + 16 + koff);
      s0 = __builtin_amdgcn_wmma_f32_16x16x32_bf16(false, aq[kk], false, f0.v, (short)0, s0, false, false);
      s1 = __builtin_amdgcn_wmma_f32_16x16x32_bf16(false, aq[kk], false, f1.v, (short)0, s1, false, false);
    }

    // ---- scale + mask + online softmax (cross-lane over 16-lane groups) ----
#pragma unroll
    for (int rr = 0; rr < 8; ++rr) {
      const int m = q0 + rr + 8 * g;
      const float* mp = mask + ((size_t)b * Sv + m) * Sv + t0;
      float v0 = s0[rr] * 0.125f + mp[r];        // 1/sqrt(64)
      float v1 = s1[rr] * 0.125f + mp[16 + r];
      float mx = fmaxf(v0, v1);
#pragma unroll
      for (int o = 8; o >= 1; o >>= 1) mx = fmaxf(mx, __shfl_xor(mx, o, 32));
      const float mnew  = fmaxf(mrow[rr], mx);
      const float alpha = __expf(mrow[rr] - mnew);
      mrow[rr] = mnew;
      v0 = __expf(v0 - mnew);
      v1 = __expf(v1 - mnew);
      float ps = v0 + v1;
#pragma unroll
      for (int o = 8; o >= 1; o >>= 1) ps += __shfl_xor(ps, o, 32);
      lrow[rr] = lrow[rr] * alpha + ps;
#pragma unroll
      for (int ni = 0; ni < 4; ++ni) ctx[ni][rr] *= alpha;
      P[wave][rr + 8 * g][r]      = (bf16_t)v0;
      P[wave][rr + 8 * g][16 + r] = (bf16_t)v1;
    }
    __syncthreads();  // uniform; orders LDS stores vs. fragment loads

    // ---- ctx += P · V   (V^T stored [d, t] row-major) ----
    Frag16 pf;
    pf.h[0] = *(const v8bf*)&P[wave][r][koff];
    pf.h[1] = *(const v8bf*)&P[wave][r][16 + koff];
#pragma unroll
    for (int ni = 0; ni < 4; ++ni) {
      const bf16_t* vrow = vtb + ((size_t)bh * HDv + (ni * 16 + r)) * Sv + t0;
      Frag16 vf;
      vf.h[0] = *(const v8bf*)(vrow + koff);
      vf.h[1] = *(const v8bf*)(vrow + 16 + koff);
      ctx[ni] = __builtin_amdgcn_wmma_f32_16x16x32_bf16(false, pf.v, false, vf.v, (short)0, ctx[ni], false, false);
    }
    __syncthreads();
  }

  // ---- normalize + write bf16 ctx [B,S,E] (heads re-interleaved) ----
#pragma unroll
  for (int ni = 0; ni < 4; ++ni)
#pragma unroll
    for (int rr = 0; rr < 8; ++rr) {
      const int m = q0 + rr + 8 * g;
      const float v = ctx[ni][rr] / lrow[rr];
      ctxb[((size_t)b * Sv + m) * Ev + h * HDv + ni * 16 + r] = (bf16_t)v;
    }
}

// ---------------------------------------------------------------------------
// out = LayerNorm(a + res) with std = sqrt(var) + eps (matches reference).
// One block per row of E=1024; optionally also emits bf16 copy.
// ---------------------------------------------------------------------------
__global__ __launch_bounds__(256) void ln_residual(
    const float* __restrict__ a, const float* __restrict__ res,
    const float* __restrict__ gamma, const float* __restrict__ beta,
    float* __restrict__ out_f, bf16_t* __restrict__ out_b) {
  const int row = blockIdx.x;
  const int t   = threadIdx.x;
  const float* pa = a + (size_t)row * Ev;
  const float* pr = res + (size_t)row * Ev;
  __shared__ float sred[8];

  float x[4];
  float s = 0.0f;
#pragma unroll
  for (int i = 0; i < 4; ++i) { x[i] = pa[t + i * 256] + pr[t + i * 256]; s += x[i]; }
#pragma unroll
  for (int o = 16; o >= 1; o >>= 1) s += __shfl_xor(s, o, 32);
  if ((t & 31) == 0) sred[t >> 5] = s;
  __syncthreads();
  float tot = 0.0f;
#pragma unroll
  for (int i = 0; i < 8; ++i) tot += sred[i];
  const float mean = tot * (1.0f / Ev);
  __syncthreads();

  float v = 0.0f;
#pragma unroll
  for (int i = 0; i < 4; ++i) { const float d = x[i] - mean; v += d * d; }
#pragma unroll
  for (int o = 16; o >= 1; o >>= 1) v += __shfl_xor(v, o, 32);
  if ((t & 31) == 0) sred[t >> 5] = v;
  __syncthreads();
  float vt = 0.0f;
#pragma unroll
  for (int i = 0; i < 8; ++i) vt += sred[i];
  const float inv = 1.0f / (sqrtf(vt * (1.0f / Ev)) + EPSv);

#pragma unroll
  for (int i = 0; i < 4; ++i) {
    const int c = t + i * 256;
    const float y = gamma[c] * ((x[i] - mean) * inv) + beta[c];
    out_f[(size_t)row * Ev + c] = y;
    if (out_b) out_b[(size_t)row * Ev + c] = (bf16_t)y;
  }
}

// ---------------------------------------------------------------------------
// Host launcher
// ---------------------------------------------------------------------------
extern "C" void kernel_launch(void* const* d_in, const int* in_sizes, int n_in,
                              void* d_out, int out_size, void* d_ws, size_t ws_size,
                              hipStream_t stream) {
  (void)in_sizes; (void)n_in; (void)out_size; (void)ws_size;
  const float* x      = (const float*)d_in[0];
  const float* mask   = (const float*)d_in[1];
  const float* Wqkv   = (const float*)d_in[2];
  const float* bqkv   = (const float*)d_in[3];
  const float* Wo     = (const float*)d_in[4];
  const float* bo     = (const float*)d_in[5];
  const float* W1     = (const float*)d_in[6];
  const float* b1     = (const float*)d_in[7];
  const float* W2     = (const float*)d_in[8];
  const float* b2     = (const float*)d_in[9];
  const float* gamma1 = (const float*)d_in[10];
  const float* beta1  = (const float*)d_in[11];
  const float* gamma2 = (const float*)d_in[12];
  const float* beta2  = (const float*)d_in[13];

  const long BS = (long)Bv * Sv;  // 8192 rows
  char*  ws  = (char*)d_ws;
  size_t off = 0;
  auto alloc = [&](size_t bytes) -> void* {
    void* p = ws + off;
    off += (bytes + 255) & ~(size_t)255;
    return p;
  };

  bf16_t* xb    = (bf16_t*)alloc((size_t)BS * Ev * 2);
  bf16_t* Wqkvb = (bf16_t*)alloc((size_t)3 * Ev * Ev * 2);
  bf16_t* Wob   = (bf16_t*)alloc((size_t)Ev * Ev * 2);
  bf16_t* W1b   = (bf16_t*)alloc((size_t)HIDv * Ev * 2);
  bf16_t* W2b   = (bf16_t*)alloc((size_t)Ev * HIDv * 2);
  float*  qkv   = (float*)alloc((size_t)BS * 3 * Ev * 4);
  bf16_t* qbuf  = (bf16_t*)alloc((size_t)BS * Ev * 2);
  bf16_t* kbuf  = (bf16_t*)alloc((size_t)BS * Ev * 2);
  bf16_t* vtbuf = (bf16_t*)alloc((size_t)BS * Ev * 2);
  bf16_t* ctxb  = (bf16_t*)alloc((size_t)BS * Ev * 2);
  float*  attno = (float*)alloc((size_t)BS * Ev * 4);
  float*  h1    = (float*)alloc((size_t)BS * Ev * 4);
  bf16_t* h1b   = (bf16_t*)alloc((size_t)BS * Ev * 2);
  bf16_t* ff1b  = (bf16_t*)alloc((size_t)BS * HIDv * 2);
  float*  ff2   = (float*)alloc((size_t)BS * Ev * 4);

  // 1) fp32 -> bf16 for x and weights
  cvt_f32_to_bf16<<<2048, 256, 0, stream>>>(x,    xb,    BS * Ev);
  cvt_f32_to_bf16<<<2048, 256, 0, stream>>>(Wqkv, Wqkvb, (long)3 * Ev * Ev);
  cvt_f32_to_bf16<<<512,  256, 0, stream>>>(Wo,   Wob,   (long)Ev * Ev);
  cvt_f32_to_bf16<<<2048, 256, 0, stream>>>(W1,   W1b,   (long)HIDv * Ev);
  cvt_f32_to_bf16<<<2048, 256, 0, stream>>>(W2,   W2b,   (long)Ev * HIDv);

  // 2) QKV projection
  gemm_bf16_wmma<false, false><<<dim3(3 * Ev / 128, BS / 128), 256, 0, stream>>>(
      xb, Wqkvb, bqkv, qkv, (int)BS, 3 * Ev, Ev);

  // 3) split heads (and transpose V)
  split_qkv_kernel<<<4096, 256, 0, stream>>>(qkv, qbuf, kbuf, vtbuf);

  // 4) attention (flash-style, WMMA)
  attention_kernel<<<dim3(Sv / 64, Bv * Hv), 128, 0, stream>>>(qbuf, kbuf, vtbuf, mask, ctxb);

  // 5) output projection + residual LN1
  gemm_bf16_wmma<false, false><<<dim3(Ev / 128, BS / 128), 256, 0, stream>>>(
      ctxb, Wob, bo, attno, (int)BS, Ev, Ev);
  ln_residual<<<(int)BS, 256, 0, stream>>>(attno, x, gamma1, beta1, h1, h1b);

  // 6) FFN
  gemm_bf16_wmma<true, true><<<dim3(HIDv / 128, BS / 128), 256, 0, stream>>>(
      h1b, W1b, b1, ff1b, (int)BS, HIDv, Ev);
  gemm_bf16_wmma<false, false><<<dim3(Ev / 128, BS / 128), 256, 0, stream>>>(
      ff1b, W2b, b2, ff2, (int)BS, Ev, HIDv);

  // 7) residual LN2 -> output
  ln_residual<<<(int)BS, 256, 0, stream>>>(ff2, h1, gamma2, beta2, (float*)d_out, (bf16_t*)nullptr);
}
